// BinaryConv2d_31353261261102
// MI455X (gfx1250) — matrix operand order repaired and checked by reading, not compiled, not run on hardware
//
#include <hip/hip_runtime.h>

typedef __attribute__((ext_vector_type(16))) _Float16 v16h;
typedef __attribute__((ext_vector_type(8)))  float    v8f;

#define N_IMG 32
#define C_IN  256
#define C_OUT 256
#define H_DIM 56
#define W_DIM 56
#define HW    3136          // 56*56
#define NPIX  (N_IMG * HW)  // 100352
#define KTOT  2304          // 9 taps * 256 ci
#define KCHUNKS 72          // 2304 / 32
#define NCOTILE 16          // 256 co / 16
// packed A: [co-tile 16][chunk 72][lane 32] x 32B fragment rows
#define APACK_FRAGS (NCOTILE * KCHUNKS * 32)
#define APACK_BYTES ((size_t)APACK_FRAGS * 32)

// ---------------------------------------------------------------------------
// Prep: binarize weights (f32 -> ±1 f16) into fragment-linear order so the
// main loop's A operand is a direct per-lane 32B global load (no per-chunk
// binarize VALU, no A panel in LDS).
// elem(mt, kc, lane, j): co = mt*16 + (lane&15); k = 16*(lane>>4) + j (contiguous K).
__global__ __launch_bounds__(256)
void pack_weights(const float* __restrict__ wgt, v16h* __restrict__ apack)
{
    const int gid  = blockIdx.x * 256 + threadIdx.x;   // 36864 fragments
    const int lane = gid & 31;
    const int rest = gid >> 5;          // mt*72 + kc
    const int kc   = rest % KCHUNKS;
    const int mt   = rest / KCHUNKS;
    const int lrow = lane & 15;
    const int fkl  = lane >> 4;

    const int tap    = kc >> 3;
    const int cibase = (kc & 7) << 5;
    const float* wRow = wgt + (size_t)(mt * 16 + lrow) * KTOT;

    v16h av;
#pragma unroll
    for (int j = 0; j < 16; ++j) {
        const int kk = (fkl << 4) + j;   // contiguous K, matches B panel
        const float wv = wRow[(cibase + kk) * 9 + tap];
        av[j] = (_Float16)(wv >= 0.0f ? 1.0f : -1.0f);
    }
    apack[gid] = av;
}

// ---------------------------------------------------------------------------
// Fast path: 256(co) x 128(pix) macro-tile per WG, K-chunk 32, 8 waves.
// A fragments streamed from packed global (L2-resident), B panel through
// double-buffered LDS with one barrier per chunk. 16 WMMA per wave per chunk.
__global__ __launch_bounds__(256)
void bconv_wmma_fast(const float* __restrict__ x,
                     const v16h*  __restrict__ apack,
                     const float* __restrict__ bias,
                     float* __restrict__ out)
{
    __shared__ v16h Bsv[2][128 * 2];   // x patch (f16), contiguous K, 16KB

    const int t    = threadIdx.x;
    const int lane = t & 31;
    const int wid  = t >> 5;
    const int wm   = wid & 3;   // 4 waves along co: 64 co (4 tiles) each
    const int wn   = wid >> 2;  // 2 waves along pix: 64 pix (4 tiles) each
    const int lrow = lane & 15;
    const int lhi  = lane >> 4;

    const int fl = t & 127;     // B fill: local pixel row
    const int fk = t >> 7;      // B fill: which 16-wide k half

    const int pix0 = blockIdx.x * 128;

    const int pB   = pix0 + fl;
    const int nB   = pB / HW;
    const int remB = pB - nB * HW;
    const int hB   = remB / W_DIM;
    const int wB   = remB - hB * W_DIM;
    const float* xImg = x + (size_t)nB * C_IN * HW;

    auto load_b = [&](int kc, v16h& bv) {
        const int tap    = kc >> 3;
        const int cibase = (kc & 7) << 5;
        const int dy     = tap / 3 - 1;
        const int dx     = tap % 3 - 1;
        const int hh = hB + dy;
        const int ww = wB + dx;
        const bool ok = (hh >= 0) && (hh < H_DIM) && (ww >= 0) && (ww < W_DIM);
        const float* xp = xImg + (size_t)(cibase + fk * 16) * HW + hh * W_DIM + ww;
#pragma unroll
        for (int i = 0; i < 16; ++i)
            bv[i] = ok ? (_Float16)xp[(size_t)i * HW] : (_Float16)0.0f;
        if (ok && (kc & 7) < 7)
            __builtin_prefetch(xp + (size_t)32 * HW, 0, 1);
    };

    v8f acc[4][4];
    const v8f vzero = {0.f, 0.f, 0.f, 0.f, 0.f, 0.f, 0.f, 0.f};
#pragma unroll
    for (int m2 = 0; m2 < 4; ++m2)
#pragma unroll
        for (int n2 = 0; n2 < 4; ++n2)
            acc[m2][n2] = vzero;

    v16h bv;
    load_b(0, bv);

    for (int kc = 0; kc < KCHUNKS; ++kc) {
        const int buf = kc & 1;

        Bsv[buf][fl * 2 + fk] = bv;
        __syncthreads();   // single barrier per chunk (double-buffered B)

        if (kc + 1 < KCHUNKS)
            load_b(kc + 1, bv);   // next chunk's x loads fly behind the WMMAs

        // A fragments: direct packed-global loads (2x b128 per fragment)
        v16h afrag[4];
#pragma unroll
        for (int m2 = 0; m2 < 4; ++m2)
            afrag[m2] = apack[((size_t)(wm * 4 + m2) * KCHUNKS + kc) * 32 + lane];

        // B fragments: one 32B LDS vector read per tile per lane
        v16h bfrag[4];
#pragma unroll
        for (int n2 = 0; n2 < 4; ++n2)
            bfrag[n2] = Bsv[buf][((wn * 4 + n2) * 16 + lrow) * 2 + lhi];

#pragma unroll
        for (int m2 = 0; m2 < 4; ++m2)
#pragma unroll
            for (int n2 = 0; n2 < 4; ++n2)
                acc[m2][n2] = __builtin_amdgcn_wmma_f32_16x16x32_f16(
                    false, afrag[m2], false, bfrag[n2],
                    (short)0, acc[m2][n2], false, false);
    }

    // epilogue: C layout = VGPR r, lanes 0-15 -> M=r, lanes 16-31 -> M=r+8
#pragma unroll
    for (int m2 = 0; m2 < 4; ++m2) {
        const int cobase = wm * 64 + m2 * 16 + lhi * 8;
        float bsgn[8];
#pragma unroll
        for (int r = 0; r < 8; ++r)
            bsgn[r] = bias[cobase + r] >= 0.0f ? 1.0f : -1.0f;
#pragma unroll
        for (int n2 = 0; n2 < 4; ++n2) {
            const int p   = pix0 + wn * 64 + n2 * 16 + lrow;
            const int ni  = p / HW;
            const int rem = p - ni * HW;
            float* op = out + (size_t)ni * C_OUT * HW + rem;
#pragma unroll
            for (int r = 0; r < 8; ++r)
                op[(size_t)(cobase + r) * HW] = acc[m2][n2][r] + bsgn[r];
        }
    }
}

// ---------------------------------------------------------------------------
// Fallback (workspace too small): self-contained 128x128 version (round-2).
__global__ __launch_bounds__(256)
void bconv_wmma_fallback(const float* __restrict__ x,
                         const float* __restrict__ wgt,
                         const float* __restrict__ bias,
                         float* __restrict__ out)
{
    __shared__ v16h Asv[2][128 * 2];
    __shared__ v16h Bsv[2][128 * 2];

    const int t    = threadIdx.x;
    const int lane = t & 31;
    const int wid  = t >> 5;
    const int wm   = wid & 3;
    const int wn   = wid >> 2;
    const int lrow = lane & 15;
    const int lhi  = lane >> 4;
    const int fl = t & 127;
    const int fk = t >> 7;

    const int pix0 = blockIdx.x * 128;
    const int co0  = blockIdx.y * 128;

    const int pB   = pix0 + fl;
    const int nB   = pB / HW;
    const int remB = pB - nB * HW;
    const int hB   = remB / W_DIM;
    const int wB   = remB - hB * W_DIM;
    const float* xImg = x + (size_t)nB * C_IN * HW;
    const float* wRow = wgt + (size_t)(co0 + fl) * KTOT;

    auto load_regs = [&](int kc, v16h& av, v16h& bv) {
        const int tap    = kc >> 3;
        const int cibase = (kc & 7) << 5;
        const int dy     = tap / 3 - 1;
        const int dx     = tap % 3 - 1;
#pragma unroll
        for (int j = 0; j < 16; ++j) {
            const int kk = (fk << 4) + j;
            const float wv = wRow[(cibase + kk) * 9 + tap];
            av[j] = (_Float16)(wv >= 0.0f ? 1.0f : -1.0f);
        }
        const int hh = hB + dy;
        const int ww = wB + dx;
        const bool ok = (hh >= 0) && (hh < H_DIM) && (ww >= 0) && (ww < W_DIM);
        const float* xp = xImg + (size_t)(cibase + fk * 16) * HW + hh * W_DIM + ww;
#pragma unroll
        for (int i = 0; i < 16; ++i)
            bv[i] = ok ? (_Float16)xp[(size_t)i * HW] : (_Float16)0.0f;
    };

    v8f acc[2][4];
    const v8f vzero = {0.f, 0.f, 0.f, 0.f, 0.f, 0.f, 0.f, 0.f};
#pragma unroll
    for (int m2 = 0; m2 < 2; ++m2)
#pragma unroll
        for (int n2 = 0; n2 < 4; ++n2)
            acc[m2][n2] = vzero;

    v16h av, bv;
    load_regs(0, av, bv);

    for (int kc = 0; kc < KCHUNKS; ++kc) {
        const int buf = kc & 1;
        Asv[buf][fl * 2 + fk] = av;
        Bsv[buf][fl * 2 + fk] = bv;
        __syncthreads();
        if (kc + 1 < KCHUNKS)
            load_regs(kc + 1, av, bv);

        v16h afrag[2], bfrag[4];
#pragma unroll
        for (int m2 = 0; m2 < 2; ++m2)
            afrag[m2] = Asv[buf][((wm * 2 + m2) * 16 + lrow) * 2 + lhi];
#pragma unroll
        for (int n2 = 0; n2 < 4; ++n2)
            bfrag[n2] = Bsv[buf][((wn * 4 + n2) * 16 + lrow) * 2 + lhi];

#pragma unroll
        for (int m2 = 0; m2 < 2; ++m2)
#pragma unroll
            for (int n2 = 0; n2 < 4; ++n2)
                acc[m2][n2] = __builtin_amdgcn_wmma_f32_16x16x32_f16(
                    false, afrag[m2], false, bfrag[n2],
                    (short)0, acc[m2][n2], false, false);
    }

#pragma unroll
    for (int m2 = 0; m2 < 2; ++m2) {
        const int cobase = co0 + wm * 32 + m2 * 16 + lhi * 8;
        float bsgn[8];
#pragma unroll
        for (int r = 0; r < 8; ++r)
            bsgn[r] = bias[cobase + r] >= 0.0f ? 1.0f : -1.0f;
#pragma unroll
        for (int n2 = 0; n2 < 4; ++n2) {
            const int p   = pix0 + wn * 64 + n2 * 16 + lrow;
            const int ni  = p / HW;
            const int rem = p - ni * HW;
            float* op = out + (size_t)ni * C_OUT * HW + rem;
#pragma unroll
            for (int r = 0; r < 8; ++r)
                op[(size_t)(cobase + r) * HW] = acc[m2][n2][r] + bsgn[r];
        }
    }
}

extern "C" void kernel_launch(void* const* d_in, const int* in_sizes, int n_in,
                              void* d_out, int out_size, void* d_ws, size_t ws_size,
                              hipStream_t stream) {
    (void)in_sizes; (void)n_in; (void)out_size;
    const float* x    = (const float*)d_in[0];
    const float* wgt  = (const float*)d_in[1];
    const float* bias = (const float*)d_in[2];
    float* out        = (float*)d_out;

    if (ws_size >= APACK_BYTES) {
        v16h* apack = (v16h*)d_ws;
        pack_weights<<<APACK_FRAGS / 256, dim3(256), 0, stream>>>(wgt, apack);
        bconv_wmma_fast<<<dim3(NPIX / 128), dim3(256), 0, stream>>>(x, apack, bias, out);
    } else {
        bconv_wmma_fallback<<<dim3(NPIX / 128, C_OUT / 128), dim3(256), 0, stream>>>(
            x, wgt, bias, out);
    }
}
